// Feedforward_56693568307221
// MI455X (gfx1250) — matrix operand order repaired and checked by reading, not compiled, not run on hardware
//
#include <hip/hip_runtime.h>
#include <math.h>

// ---------------------------------------------------------------------------
// GRU LM forward for MI455X (gfx1250, wave32, WMMA).
//
// Sizes (fixed by reference): B=512, T=512, V=256, E=256, H=256, 3H=768.
//
// Strategy:
//  * Fold embedding + input projection into a V x 3H lookup table:
//        table[v] = wte[v] @ w_ih.T + b_ih          (kills ~103 GFLOP + 384MB)
//  * Persistent recurrence kernel: 32 workgroups, each owns 16 batch rows,
//    loops T=512 steps.  Per step: gh = h @ w_hh.T via v_wmma_f32_16x16x32_bf16
//    (f32 accumulate), then gate math in f32, h stored bf16 in LDS.
//  * w_hh / w_out pre-packed in B-fragment layout (bf16), streamed from L2
//    (393KB, fully L2-resident; no intra-WG reuse, so LDS staging would lose).
//  * Final logits GEMM with the same WMMA path, bias pre-seeded into C.
// ---------------------------------------------------------------------------

typedef __attribute__((ext_vector_type(16))) __bf16 v16bf;
typedef __attribute__((ext_vector_type(8)))  float  v8f;

constexpr int Bsz = 512, Tsz = 512, Vsz = 256, Esz = 256, Hsz = 256;
constexpr int G3  = 3 * Hsz;           // 768
constexpr int BT  = 16;                // batch rows per workgroup
constexpr int NWAVES = 8;              // 256 threads = 8 wave32
constexpr int KT  = Hsz / 32;          // 8 K-tiles (K depth 32 per WMMA)
constexpr int NT_HH  = G3 / 16;        // 48 N-tiles for the recurrence GEMM
constexpr int NT_OUT = Vsz / 16;       // 16 N-tiles for the logits GEMM
constexpr int HPAD = Hsz + 8;          // padded LDS row (bf16) -> conflict-free b128 reads

// ---------------------------------------------------------------------------
// table[v][g] = sum_e wte[v][e] * w_ih[g][e] + b_ih[g]        (one-time, tiny)
// ---------------------------------------------------------------------------
__global__ void build_table(const float* __restrict__ wte,
                            const float* __restrict__ w_ih,
                            const float* __restrict__ b_ih,
                            float* __restrict__ table) {
  int t = blockIdx.x * blockDim.x + threadIdx.x;
  if (t >= Vsz * G3) return;
  int g = t % G3, v = t / G3;
  float acc = b_ih[g];
  const float* wr = wte  + (size_t)v * Esz;
  const float* gr = w_ih + (size_t)g * Esz;
#pragma unroll 8
  for (int e = 0; e < Esz; ++e) acc = fmaf(wr[e], gr[e], acc);
  table[(size_t)v * G3 + g] = acc;
}

// ---------------------------------------------------------------------------
// Pack W[N][K] (row-major f32) into bf16 WMMA B-fragments for 16x16x32:
//   B element (k,n): n = lane%16, k = ktile*32 + (lane/16)*16 + 2v + half.
// One thread per dword (bf16 pair).
// ---------------------------------------------------------------------------
__global__ void pack_bfrag(const float* __restrict__ W, __bf16* __restrict__ dst,
                           int ntiles, int ktiles, int ldw) {
  int t = blockIdx.x * blockDim.x + threadIdx.x;
  int total = ktiles * ntiles * 32 * 8;
  if (t >= total) return;
  int v    = t & 7;
  int lane = (t >> 3) & 31;
  int blk  = t >> 8;                 // kt*ntiles + nt
  int nt   = blk % ntiles;
  int kt   = blk / ntiles;
  int n = nt * 16 + (lane & 15);
  int k = kt * 32 + (lane >> 4) * 16 + v * 2;
  __bf16* d = dst + ((size_t)blk * 32 + lane) * 16 + v * 2;
  d[0] = (__bf16)W[(size_t)n * ldw + k];
  d[1] = (__bf16)W[(size_t)n * ldw + k + 1];
}

// ---------------------------------------------------------------------------
// Persistent GRU recurrence + logits.  grid = B/BT = 32, block = 256 (8 waves).
// ---------------------------------------------------------------------------
__global__ void __launch_bounds__(256)
gru_persistent(const int*   __restrict__ g_idx,
               const float* __restrict__ table,
               const v16bf* __restrict__ whhB,
               const v16bf* __restrict__ woutB,
               const float* __restrict__ b_hh,
               const float* __restrict__ b_out,
               float* __restrict__ out) {
  __shared__ __bf16 h_lds[BT][HPAD];      // hidden state, bf16
  __shared__ float  gh_lds[BT][G3];       // per-step recurrent projections, f32

  const int tid  = threadIdx.x;
  const int lane = tid & 31;
  const int wave = tid >> 5;
  const int lo   = lane & 15;
  const int hi   = lane >> 4;
  const int rowBase = blockIdx.x * BT;

  // h0 = 0
#pragma unroll
  for (int i = 0; i < BT; ++i) h_lds[i][tid] = (__bf16)0.0f;
  __syncthreads();

  // gate-phase biases (thread tid owns hidden column j = tid)
  const float bh_r = b_hh[tid];
  const float bh_z = b_hh[tid + Hsz];
  const float bh_n = b_hh[tid + 2 * Hsz];

  for (int t = 0; t < Tsz; ++t) {
    // ---- phase 1: gh = h @ w_hh.T  (WMMA bf16, f32 accumulate) ----
    v16bf a[KT];
#pragma unroll
    for (int kt = 0; kt < KT; ++kt) {
      // A fragment (16x32 bf16): m = lane%16, element s: K = hi*16 + s (matches B packing)
      const __bf16* base = &h_lds[lo][kt * 32 + hi * 16];
      union { v16bf v; uint4 q[2]; } u;
      u.q[0] = *(const uint4*)(base);        // K = hi*16 .. hi*16+7
      u.q[1] = *(const uint4*)(base + 8);    // K = hi*16+8 .. hi*16+15
      a[kt] = u.v;
    }
#pragma unroll
    for (int i = 0; i < NT_HH / NWAVES; ++i) {
      const int nt = wave * (NT_HH / NWAVES) + i;
      v8f c = {0.f, 0.f, 0.f, 0.f, 0.f, 0.f, 0.f, 0.f};
#pragma unroll
      for (int kt = 0; kt < KT; ++kt) {
        v16bf b = whhB[((size_t)(kt * NT_HH + nt)) * 32 + lane];  // L2-resident stream
        c = __builtin_amdgcn_wmma_f32_16x16x32_bf16(
                false, a[kt], false, b, (short)0, c, false, false);
      }
#pragma unroll
      for (int v = 0; v < 8; ++v)
        gh_lds[v + hi * 8][nt * 16 + lo] = c[v];   // C layout: m = v + hi*8, n = lo
    }
    __syncthreads();

    // ---- phase 2: gates + hidden update (thread tid = column j, 16 rows) ----
#pragma unroll 4
    for (int i = 0; i < BT; ++i) {
      const int   tok = g_idx[(size_t)(rowBase + i) * Tsz + t];
      const float* xi = table + (size_t)tok * G3;
      const float i_r = xi[tid];
      const float i_z = xi[tid + Hsz];
      const float i_n = xi[tid + 2 * Hsz];
      const float h_r = gh_lds[i][tid]           + bh_r;
      const float h_z = gh_lds[i][tid + Hsz]     + bh_z;
      const float h_n = gh_lds[i][tid + 2 * Hsz] + bh_n;
      const float r = 1.0f / (1.0f + __expf(-(i_r + h_r)));
      const float z = 1.0f / (1.0f + __expf(-(i_z + h_z)));
      const float n = tanhf(i_n + r * h_n);
      const float hp = (float)h_lds[i][tid];
      h_lds[i][tid] = (__bf16)((1.0f - z) * n + z * hp);
    }
    __syncthreads();
  }

  // ---- relu(hT) in place ----
#pragma unroll
  for (int i = 0; i < BT; ++i) {
    float hv = (float)h_lds[i][tid];
    h_lds[i][tid] = (__bf16)(hv > 0.0f ? hv : 0.0f);
  }
  __syncthreads();

  // ---- logits = relu(h) @ w_out.T + b_out ----
  {
    v16bf a[KT];
#pragma unroll
    for (int kt = 0; kt < KT; ++kt) {
      const __bf16* base = &h_lds[lo][kt * 32 + hi * 16];
      union { v16bf v; uint4 q[2]; } u;
      u.q[0] = *(const uint4*)(base);
      u.q[1] = *(const uint4*)(base + 8);
      a[kt] = u.v;
    }
#pragma unroll
    for (int i = 0; i < NT_OUT / NWAVES; ++i) {
      const int nt = wave * (NT_OUT / NWAVES) + i;
      const float bias = b_out[nt * 16 + lo];
      v8f c;
#pragma unroll
      for (int v = 0; v < 8; ++v) c[v] = bias;
#pragma unroll
      for (int kt = 0; kt < KT; ++kt) {
        v16bf b = woutB[((size_t)(kt * NT_OUT + nt)) * 32 + lane];
        c = __builtin_amdgcn_wmma_f32_16x16x32_bf16(
                false, a[kt], false, b, (short)0, c, false, false);
      }
#pragma unroll
      for (int v = 0; v < 8; ++v)
        out[(size_t)(rowBase + v + hi * 8) * Vsz + nt * 16 + lo] = c[v];
    }
  }
}

// ---------------------------------------------------------------------------
extern "C" void kernel_launch(void* const* d_in, const int* in_sizes, int n_in,
                              void* d_out, int out_size, void* d_ws, size_t ws_size,
                              hipStream_t stream) {
  (void)in_sizes; (void)n_in; (void)out_size; (void)ws_size;
  const int*   idx   = (const int*)  d_in[0];
  const float* wte   = (const float*)d_in[1];
  const float* w_ih  = (const float*)d_in[2];
  const float* w_hh  = (const float*)d_in[3];
  const float* b_ih  = (const float*)d_in[4];
  const float* b_hh  = (const float*)d_in[5];
  const float* w_out = (const float*)d_in[6];
  const float* b_out = (const float*)d_in[7];
  float* out = (float*)d_out;

  // workspace layout (all offsets 32B-aligned)
  char* ws = (char*)d_ws;
  const size_t table_bytes = (size_t)Vsz * G3 * sizeof(float);                 // 768 KB
  const size_t whh_bytes   = (size_t)KT * NT_HH * 32 * 16 * sizeof(__bf16);    // 384 KB
  float*  table = (float*)(ws);
  __bf16* whhB  = (__bf16*)(ws + table_bytes);
  __bf16* woutB = (__bf16*)(ws + table_bytes + whh_bytes);

  // 1) fold embedding + input projection into the V x 3H table
  build_table<<<(Vsz * G3 + 255) / 256, 256, 0, stream>>>(wte, w_ih, b_ih, table);
  // 2) pack weights into WMMA B-fragment layout (bf16)
  pack_bfrag<<<(KT * NT_HH * 32 * 8 + 255) / 256, 256, 0, stream>>>(
      w_hh, whhB, NT_HH, KT, Hsz);
  pack_bfrag<<<(KT * NT_OUT * 32 * 8 + 255) / 256, 256, 0, stream>>>(
      w_out, woutB, NT_OUT, KT, Hsz);
  // 3) persistent recurrence + logits
  gru_persistent<<<Bsz / BT, 256, 0, stream>>>(
      idx, table, (const v16bf*)whhB, (const v16bf*)woutB, b_hh, b_out, out);
}
